// ParallelTransformerBlock_15977278341454
// MI455X (gfx1250) — matrix operand order repaired and checked by reading, not compiled, not run on hardware
//
#include <hip/hip_runtime.h>
#include <hip/hip_bf16.h>

// ---------------------------------------------------------------------------
// Problem constants (from the reference)
// ---------------------------------------------------------------------------
constexpr int Bc   = 4;
constexpr int Nc   = 4096;
constexpr int Dc   = 1024;
constexpr int Hc   = 16;
constexpr int DHc  = 64;
constexpr int SEGc = 64;
constexpr int Gc   = Nc / SEGc;      // 64
constexpr int DFFc = 4 * Dc;         // 4096
constexpr int BNTOK = Bc * Nc;       // 16384 rows

// ---------------------------------------------------------------------------
// WMMA / TDM types (CDNA5 / gfx1250, wave32)
// ---------------------------------------------------------------------------
typedef __attribute__((ext_vector_type(16))) __bf16        v16bf;
typedef __attribute__((ext_vector_type(8)))  float         v8f;
typedef __attribute__((ext_vector_type(4)))  unsigned int  v4u;
typedef __attribute__((ext_vector_type(8)))  int           v8i;
typedef __attribute__((ext_vector_type(4)))  int           v4i;

union FragAB {
    unsigned int u[8];
    v16bf        v;
};

__device__ __forceinline__ unsigned short f32_to_bf16(float f) {
    unsigned int u = __float_as_uint(f);
    unsigned int r = u + 0x7FFFu + ((u >> 16) & 1u);   // round-to-nearest-even
    return (unsigned short)(r >> 16);
}

// A-fragment (16x32 bf16): pairs at k = 2p + 16*half
__device__ __forceinline__ void load_fragA(FragAB& f, const unsigned short* row,
                                           int half_) {
#pragma unroll
    for (int p = 0; p < 8; ++p)
        f.u[p] = *(const unsigned int*)(row + 2 * p + half_ * 16);
}
// B-fragment (32x16 bf16): pairs at k = 2p + 16*half
__device__ __forceinline__ void load_fragB(FragAB& f, const unsigned short* row,
                                           int half_) {
#pragma unroll
    for (int p = 0; p < 8; ++p)
        f.u[p] = *(const unsigned int*)(row + 2 * p + half_ * 16);
}

// ---------------------------------------------------------------------------
// Tensor Data Mover: async 2D tile load (rows x 64 bf16) global -> LDS.
// Row in LDS padded: 32 dwords data + 4 dwords pad = 72 bf16 stride.
// Descriptor packing per CDNA5 ISA.
// ---------------------------------------------------------------------------
__device__ __forceinline__ unsigned lds_off32(const void* p) {
    return (unsigned)(unsigned long long)(size_t)p;   // low 32 bits = LDS offset
}

__device__ __forceinline__ void tdm_load_tile(const unsigned short* gptr,
                                              unsigned lds_addr, int rows,
                                              int strideElems) {
    unsigned long long ga = (unsigned long long)(size_t)gptr;
    v4u g0;
    g0[0] = 1u;                                            // count=1 (valid, user)
    g0[1] = lds_addr;                                      // LDS byte address
    g0[2] = (unsigned)(ga & 0xFFFFFFFFu);                  // global_addr[31:0]
    g0[3] = (unsigned)((ga >> 32) & 0x01FFFFFFu) | (2u << 30);  // [56:32] | type=2
    const unsigned rowDw = 32u;                            // 64 bf16 = 32 dwords
    const unsigned strideDw = (unsigned)(strideElems >> 1);
    v8i g1;
    g1[0] = (int)((2u << 16) |                             // data_size = 4 B
                  (1u << 20) |                             // pad_enable
                  (4u << 22) |                             // pad_interval: 32 dw
                  (3u << 25));                             // pad_amount: 4 dw
    g1[1] = (int)(rowDw << 16);                            // tensor_dim0[15:0]
    g1[2] = (int)(((unsigned)rows & 0xFFFFu) << 16);       // tensor_dim1[15:0]
    g1[3] = (int)(rowDw << 16);                            // tile_dim0
    g1[4] = (int)((unsigned)rows & 0xFFFFu);               // tile_dim1 (tile_dim2=0)
    g1[5] = (int)strideDw;                                 // tensor_dim0_stride lo
    g1[6] = 0;                                             // stride hi / dim1_stride
    g1[7] = 0;
    v4i z = {0, 0, 0, 0};
#if defined(__clang_major__) && (__clang_major__ >= 23)
    v8i z8 = {0, 0, 0, 0, 0, 0, 0, 0};
    __builtin_amdgcn_tensor_load_to_lds(g0, g1, z, z, z8, 0);
#else
    __builtin_amdgcn_tensor_load_to_lds(g0, g1, z, z, 0);
#endif
}

// ---------------------------------------------------------------------------
// float32 -> bf16 transposed weight conversion: W[K,N] -> WT[N,K]
// ---------------------------------------------------------------------------
__global__ __launch_bounds__(256) void cvt_transpose_bf16(
    const float* __restrict__ w, unsigned short* __restrict__ wt, int K, int Nn) {
    __shared__ unsigned short tile[32][33];
    const int nt = blockIdx.x * 32, kt = blockIdx.y * 32;
    const int tx = threadIdx.x & 31, ty = threadIdx.x >> 5;   // ty 0..7
#pragma unroll
    for (int j = 0; j < 4; ++j)
        tile[ty + 8 * j][tx] = f32_to_bf16(w[(size_t)(kt + ty + 8 * j) * Nn + nt + tx]);
    __syncthreads();
#pragma unroll
    for (int j = 0; j < 4; ++j)
        wt[(size_t)(nt + ty + 8 * j) * K + kt + tx] = tile[tx][ty + 8 * j];
}

// ---------------------------------------------------------------------------
// Block-wide sum over 256 threads (wave32)
// ---------------------------------------------------------------------------
__device__ __forceinline__ float blk_sum(float v, float* red, int tid) {
#pragma unroll
    for (int o = 16; o > 0; o >>= 1) v += __shfl_xor(v, o, 32);
    if ((tid & 31) == 0) red[tid >> 5] = v;
    __syncthreads();
    float t = red[tid & 7];
#pragma unroll
    for (int o = 4; o > 0; o >>= 1) t += __shfl_xor(t, o, 8);
    __syncthreads();
    return t;
}

// ---------------------------------------------------------------------------
// LayerNorm(x, g, 0) -> bf16   (one block per row, D = 1024, 256 threads)
// ---------------------------------------------------------------------------
__global__ __launch_bounds__(256) void ln_kernel(const float* __restrict__ x,
                                                 const float* __restrict__ g,
                                                 unsigned short* __restrict__ out) {
    __shared__ float red[8];
    const int row = blockIdx.x, tid = threadIdx.x;
    const float4 v = ((const float4*)(x + (size_t)row * Dc))[tid];
    float s  = v.x + v.y + v.z + v.w;
    float sq = v.x * v.x + v.y * v.y + v.z * v.z + v.w * v.w;
    s  = blk_sum(s,  red, tid);
    sq = blk_sum(sq, red, tid);
    const float mean = s * (1.0f / Dc);
    const float rs   = rsqrtf(sq * (1.0f / Dc) - mean * mean + 1e-5f);
    const float4 gg  = ((const float4*)g)[tid];
    unsigned short o4[4];
    o4[0] = f32_to_bf16((v.x - mean) * rs * gg.x);
    o4[1] = f32_to_bf16((v.y - mean) * rs * gg.y);
    o4[2] = f32_to_bf16((v.z - mean) * rs * gg.z);
    o4[3] = f32_to_bf16((v.w - mean) * rs * gg.w);
    *(uint2*)(out + (size_t)row * Dc + tid * 4) = *(uint2*)o4;
}

// ---------------------------------------------------------------------------
// Double LayerNorm: LN(LN(x, g1, 0), g2, b2) -> bf16
// ---------------------------------------------------------------------------
__global__ __launch_bounds__(256) void ln2_kernel(const float* __restrict__ x,
                                                  const float* __restrict__ g1,
                                                  const float* __restrict__ g2,
                                                  const float* __restrict__ b2,
                                                  unsigned short* __restrict__ out) {
    __shared__ float red[8];
    const int row = blockIdx.x, tid = threadIdx.x;
    const float4 v = ((const float4*)(x + (size_t)row * Dc))[tid];
    float s  = blk_sum(v.x + v.y + v.z + v.w, red, tid);
    float sq = blk_sum(v.x * v.x + v.y * v.y + v.z * v.z + v.w * v.w, red, tid);
    float mean = s * (1.0f / Dc);
    float rs   = rsqrtf(sq * (1.0f / Dc) - mean * mean + 1e-5f);
    const float4 gg1 = ((const float4*)g1)[tid];
    float y0 = (v.x - mean) * rs * gg1.x;
    float y1 = (v.y - mean) * rs * gg1.y;
    float y2 = (v.z - mean) * rs * gg1.z;
    float y3 = (v.w - mean) * rs * gg1.w;
    s  = blk_sum(y0 + y1 + y2 + y3, red, tid);
    sq = blk_sum(y0 * y0 + y1 * y1 + y2 * y2 + y3 * y3, red, tid);
    mean = s * (1.0f / Dc);
    rs   = rsqrtf(sq * (1.0f / Dc) - mean * mean + 1e-5f);
    const float4 gg2 = ((const float4*)g2)[tid];
    const float4 bb2 = ((const float4*)b2)[tid];
    unsigned short o4[4];
    o4[0] = f32_to_bf16((y0 - mean) * rs * gg2.x + bb2.x);
    o4[1] = f32_to_bf16((y1 - mean) * rs * gg2.y + bb2.y);
    o4[2] = f32_to_bf16((y2 - mean) * rs * gg2.z + bb2.z);
    o4[3] = f32_to_bf16((y3 - mean) * rs * gg2.w + bb2.w);
    *(uint2*)(out + (size_t)row * Dc + tid * 4) = *(uint2*)o4;
}

// ---------------------------------------------------------------------------
// bf16 WMMA GEMM with TDM double-buffered tiles.
//   C[M,N] = A[M,K] * BT[N,K]^T + bias
//   EPI 0: bf16   EPI 1: exact GELU -> bf16   EPI 2: + residual(f32) -> f32
// Block: 256 threads = 8 waves; tile 128(M) x 128(N) x 64(K).
// Wave tile 32x64 = 2x4 WMMA tiles -> 16 v_wmma per K-iteration per wave.
// ---------------------------------------------------------------------------
constexpr int BM = 128, BN = 128, BK = 64;
constexpr int LDS_K = BK + 8;                      // 72 elems (TDM pad: 32+4 dw)
constexpr int GEMM_SMEM = 2 * (BM + BN) * LDS_K * 2;   // 73728 B (dynamic LDS)

template <int EPI>
__global__ __launch_bounds__(256) void gemm_bf16(
    const unsigned short* __restrict__ A, const unsigned short* __restrict__ BT,
    const float* __restrict__ bias, const float* __restrict__ resid,
    void* __restrict__ Cout, int M, int Nn, int K) {
    extern __shared__ __align__(16) unsigned short smem[];
    unsigned short* As = smem;                         // 2 x BM x LDS_K
    unsigned short* Bs = smem + 2 * BM * LDS_K;        // 2 x BN x LDS_K

    const int tid = threadIdx.x;
    const int m0b = blockIdx.y * BM;
    const int n0b = blockIdx.x * BN;
    const int w = tid >> 5, lane = tid & 31;
    const int wmi = (w & 3) * 32;                      // wave row offset in block
    const int wni = (w >> 2) * 64;                     // wave col offset in block
    const int lm = lane & 15, half_ = lane >> 4;

    v8f acc[2][4] = {};

    const unsigned short* Abase = A  + (size_t)m0b * K;
    const unsigned short* Bbase = BT + (size_t)n0b * K;
    const int nk = K / BK;

    if (w == 0) {                                       // wave 0 drives the TDM
        tdm_load_tile(Abase, lds_off32(As), BM, K);
        tdm_load_tile(Bbase, lds_off32(Bs), BN, K);
        __builtin_amdgcn_s_wait_tensorcnt(0);
    }
    __syncthreads();

    for (int i = 0; i < nk; ++i) {
        const unsigned short* Ac = As + (i & 1) * BM * LDS_K;
        const unsigned short* Bc = Bs + (i & 1) * BN * LDS_K;
        if (w == 0 && i + 1 < nk) {                     // prefetch next K-tile
            const int nx = (i + 1) & 1;
            tdm_load_tile(Abase + (size_t)(i + 1) * BK,
                          lds_off32(As + nx * BM * LDS_K), BM, K);
            tdm_load_tile(Bbase + (size_t)(i + 1) * BK,
                          lds_off32(Bs + nx * BN * LDS_K), BN, K);
        }
#pragma unroll
        for (int ks = 0; ks < BK; ks += 32) {
            FragAB af[2], bfr[4];
#pragma unroll
            for (int mi = 0; mi < 2; ++mi)
                load_fragA(af[mi], Ac + (wmi + mi * 16 + lm) * LDS_K + ks, half_);
#pragma unroll
            for (int ni = 0; ni < 4; ++ni)
                load_fragB(bfr[ni], Bc + (wni + ni * 16 + lm) * LDS_K + ks, half_);
#pragma unroll
            for (int mi = 0; mi < 2; ++mi)
#pragma unroll
                for (int ni = 0; ni < 4; ++ni)
                    acc[mi][ni] = __builtin_amdgcn_wmma_f32_16x16x32_bf16(
                        false, af[mi].v, false, bfr[ni].v, (short)0, acc[mi][ni],
                        false, false);
        }
        if (w == 0) __builtin_amdgcn_s_wait_tensorcnt(0);
        __syncthreads();
    }

    // ---- epilogue ----
#pragma unroll
    for (int mi = 0; mi < 2; ++mi)
#pragma unroll
        for (int ni = 0; ni < 4; ++ni) {
            const int gn = n0b + wni + ni * 16 + lm;
            const float bi = bias[gn];
#pragma unroll
            for (int r = 0; r < 8; ++r) {
                const int gm = m0b + wmi + mi * 16 + r + 8 * half_;
                float val = acc[mi][ni][r] + bi;
                const size_t off = (size_t)gm * Nn + gn;
                if (EPI == 0) {
                    ((unsigned short*)Cout)[off] = f32_to_bf16(val);
                } else if (EPI == 1) {
                    float ge = 0.5f * val * (1.0f + erff(val * 0.70710678118654752f));
                    ((unsigned short*)Cout)[off] = f32_to_bf16(ge);
                } else {
                    ((float*)Cout)[off] = val + resid[off];
                }
            }
        }
}

// ---------------------------------------------------------------------------
// Dilated attention, one block per (b, g, h), 64 threads (2 waves).
// Head h attends the 32 segment positions with parity (h & 1).
// S = Q K^T * DH^-0.5 (WMMA), softmax over 32, O = P V (WMMA).
// Also zero-fills the opposite-parity rows of this head's column block.
// ---------------------------------------------------------------------------
constexpr int AT_LDQ = DHc + 8;  // 72
constexpr int AT_LDV = 32 + 8;   // 40
constexpr int AT_LDP = 32 + 8;   // 40

__global__ __launch_bounds__(64) void attn_kernel(
    const unsigned short* __restrict__ q, const unsigned short* __restrict__ k,
    const unsigned short* __restrict__ v, unsigned short* __restrict__ o) {
    __shared__ __align__(16) unsigned short Qs[32 * AT_LDQ];
    __shared__ __align__(16) unsigned short Ks[32 * AT_LDQ];
    __shared__ __align__(16) unsigned short Vt[64 * AT_LDV];  // [dim][pos]
    __shared__ __align__(16) unsigned short Ps[32 * AT_LDP];

    const int bid = blockIdx.x;
    const int h = bid & (Hc - 1);
    const int g = (bid >> 4) & (Gc - 1);
    const int b = bid >> 10;
    const int tid = threadIdx.x;
    const int par = h & 1;

#pragma unroll
    for (int c = 0; c < 4; ++c) {
        const int lin = tid + 64 * c;
        const int rowm = lin >> 3;
        const int v8idx = lin & 7;
        const int tok = g * SEGc + par + 2 * rowm;
        const size_t goff = ((size_t)(b * Nc + tok)) * Dc + h * DHc + v8idx * 8;
        uint4 qa = *(const uint4*)(q + goff);
        uint4 ka = *(const uint4*)(k + goff);
        uint4 va = *(const uint4*)(v + goff);
        *(uint4*)&Qs[rowm * AT_LDQ + v8idx * 8] = qa;
        *(uint4*)&Ks[rowm * AT_LDQ + v8idx * 8] = ka;
        unsigned short tv[8];
        *(uint4*)tv = va;
#pragma unroll
        for (int j = 0; j < 8; ++j) Vt[(v8idx * 8 + j) * AT_LDV + rowm] = tv[j];
    }
    __syncthreads();

    const int wid = tid >> 5, lane = tid & 31;
    const int lm = lane & 15, half_ = lane >> 4;
    const int m0 = wid * 16;

    v8f s[2] = {};
#pragma unroll
    for (int kk = 0; kk < DHc; kk += 32) {
        FragAB af;
        load_fragA(af, &Qs[(m0 + lm) * AT_LDQ + kk], half_);
#pragma unroll
        for (int ni = 0; ni < 2; ++ni) {
            FragAB bf_;
            load_fragB(bf_, &Ks[(ni * 16 + lm) * AT_LDQ + kk], half_);
            s[ni] = __builtin_amdgcn_wmma_f32_16x16x32_bf16(
                false, af.v, false, bf_.v, (short)0, s[ni], false, false);
        }
    }

    const float scale = 0.125f;  // 64^-0.5
#pragma unroll
    for (int r = 0; r < 8; ++r) {
        float a0 = s[0][r] * scale, a1 = s[1][r] * scale;
        float mx = fmaxf(a0, a1);
#pragma unroll
        for (int off = 8; off > 0; off >>= 1) mx = fmaxf(mx, __shfl_xor(mx, off, 16));
        float e0 = __expf(a0 - mx), e1 = __expf(a1 - mx);
        float sm = e0 + e1;
#pragma unroll
        for (int off = 8; off > 0; off >>= 1) sm += __shfl_xor(sm, off, 16);
        float inv = 1.0f / sm;
        const int m = m0 + r + 8 * half_;
        Ps[m * AT_LDP + lm]      = f32_to_bf16(e0 * inv);
        Ps[m * AT_LDP + 16 + lm] = f32_to_bf16(e1 * inv);
    }
    __syncthreads();

    FragAB pf;
    load_fragA(pf, &Ps[(m0 + lm) * AT_LDP], half_);
    v8f oacc[4] = {};
#pragma unroll
    for (int ni = 0; ni < 4; ++ni) {
        FragAB bf_;
        load_fragB(bf_, &Vt[(ni * 16 + lm) * AT_LDV], half_);
        oacc[ni] = __builtin_amdgcn_wmma_f32_16x16x32_bf16(
            false, pf.v, false, bf_.v, (short)0, oacc[ni], false, false);
    }

#pragma unroll
    for (int ni = 0; ni < 4; ++ni) {
        const int dim = h * DHc + ni * 16 + lm;
#pragma unroll
        for (int r = 0; r < 8; ++r) {
            const int m = m0 + r + 8 * half_;
            const int tok  = g * SEGc + par + 2 * m;
            const int tok0 = g * SEGc + (par ^ 1) + 2 * m;
            o[(size_t)(b * Nc + tok)  * Dc + dim] = f32_to_bf16(oacc[ni][r]);
            o[(size_t)(b * Nc + tok0) * Dc + dim] = 0;
        }
    }
}

// ---------------------------------------------------------------------------
// Host-side orchestration
// ---------------------------------------------------------------------------
extern "C" void kernel_launch(void* const* d_in, const int* in_sizes, int n_in,
                              void* d_out, int out_size, void* d_ws, size_t ws_size,
                              hipStream_t stream) {
    (void)in_sizes; (void)n_in; (void)out_size; (void)ws_size;
    const float* x    = (const float*)d_in[0];
    const float* gmma = (const float*)d_in[1];
    const float* wq   = (const float*)d_in[2];
    const float* bq   = (const float*)d_in[3];
    const float* wk   = (const float*)d_in[4];
    const float* bk   = (const float*)d_in[5];
    const float* wv   = (const float*)d_in[6];
    const float* bv   = (const float*)d_in[7];
    const float* wo   = (const float*)d_in[8];
    const float* bo   = (const float*)d_in[9];
    const float* ff_g = (const float*)d_in[10];
    const float* ff_b = (const float*)d_in[11];
    const float* w1   = (const float*)d_in[12];
    const float* b1   = (const float*)d_in[13];
    const float* w2   = (const float*)d_in[14];
    const float* b2   = (const float*)d_in[15];

    char* ws = (char*)d_ws;
    const size_t SZ_DD  = (size_t)Dc * Dc;
    const size_t SZ_DF  = (size_t)Dc * DFFc;
    const size_t SZ_ACT = (size_t)BNTOK * Dc;
    size_t off = 0;
    auto alloc = [&](size_t bytes) { size_t o = off; off += (bytes + 255) & ~(size_t)255; return o; };
    const size_t o_wq = alloc(SZ_DD * 2);
    const size_t o_wk = alloc(SZ_DD * 2);
    const size_t o_wv = alloc(SZ_DD * 2);
    const size_t o_wo = alloc(SZ_DD * 2);
    const size_t o_w1 = alloc(SZ_DF * 2);
    const size_t o_w2 = alloc(SZ_DF * 2);
    const size_t o_x2 = alloc(SZ_ACT * 4);
    const size_t o_h2 = alloc(SZ_ACT * 2);
    const size_t o_hA = alloc(SZ_ACT * 2);
    const size_t o_q  = alloc(SZ_ACT * 2);
    const size_t o_k  = alloc(SZ_ACT * 2);
    const size_t o_v  = alloc(SZ_ACT * 2);
    (void)alloc(SZ_ACT * 2);                       // tail so FFN-mid alias fits
    unsigned short* wqT = (unsigned short*)(ws + o_wq);   // all weights stored [N][K]
    unsigned short* wkT = (unsigned short*)(ws + o_wk);
    unsigned short* wvT = (unsigned short*)(ws + o_wv);
    unsigned short* woT = (unsigned short*)(ws + o_wo);
    unsigned short* w1T = (unsigned short*)(ws + o_w1);
    unsigned short* w2T = (unsigned short*)(ws + o_w2);
    float*          x2  = (float*)(ws + o_x2);
    unsigned short* h2b = (unsigned short*)(ws + o_h2);
    unsigned short* hA  = (unsigned short*)(ws + o_hA);   // LN1 out / attn out
    unsigned short* qb  = (unsigned short*)(ws + o_q);
    unsigned short* kb2 = (unsigned short*)(ws + o_k);
    unsigned short* vb2 = (unsigned short*)(ws + o_v);
    unsigned short* mid = qb;                             // [BNTOK, DFF] alias

    // 1) weights -> transposed bf16 ([K,N] f32 -> [N,K] bf16)
    cvt_transpose_bf16<<<dim3(Dc / 32, Dc / 32),   256, 0, stream>>>(wq, wqT, Dc, Dc);
    cvt_transpose_bf16<<<dim3(Dc / 32, Dc / 32),   256, 0, stream>>>(wk, wkT, Dc, Dc);
    cvt_transpose_bf16<<<dim3(Dc / 32, Dc / 32),   256, 0, stream>>>(wv, wvT, Dc, Dc);
    cvt_transpose_bf16<<<dim3(Dc / 32, Dc / 32),   256, 0, stream>>>(wo, woT, Dc, Dc);
    cvt_transpose_bf16<<<dim3(DFFc / 32, Dc / 32), 256, 0, stream>>>(w1, w1T, Dc, DFFc);
    cvt_transpose_bf16<<<dim3(Dc / 32, DFFc / 32), 256, 0, stream>>>(w2, w2T, DFFc, Dc);

    // 2) h = LN(x, gamma, 0)
    ln_kernel<<<BNTOK, 256, 0, stream>>>(x, gmma, hA);

    // 3) q,k,v projections
    dim3 gqkv(Dc / BN, BNTOK / BM);
    gemm_bf16<0><<<gqkv, 256, GEMM_SMEM, stream>>>(hA, wqT, bq, nullptr, qb,  BNTOK, Dc, Dc);
    gemm_bf16<0><<<gqkv, 256, GEMM_SMEM, stream>>>(hA, wkT, bk, nullptr, kb2, BNTOK, Dc, Dc);
    gemm_bf16<0><<<gqkv, 256, GEMM_SMEM, stream>>>(hA, wvT, bv, nullptr, vb2, BNTOK, Dc, Dc);

    // 4) dilated attention -> hA (aliased; h no longer needed)
    attn_kernel<<<Bc * Gc * Hc, 64, 0, stream>>>(qb, kb2, vb2, hA);

    // 5) x2 = x + attn @ wo + bo
    gemm_bf16<2><<<gqkv, 256, GEMM_SMEM, stream>>>(hA, woT, bo, x, x2, BNTOK, Dc, Dc);

    // 6) h2 = LN(LN(x2, gamma, 0), ff_g, ff_b)
    ln2_kernel<<<BNTOK, 256, 0, stream>>>(x2, gmma, ff_g, ff_b, h2b);

    // 7) mid = gelu(h2 @ w1 + b1)
    dim3 gff1(DFFc / BN, BNTOK / BM);
    gemm_bf16<1><<<gff1, 256, GEMM_SMEM, stream>>>(h2b, w1T, b1, nullptr, mid, BNTOK, DFFc, Dc);

    // 8) out = x2 + mid @ w2 + b2
    dim3 gff2(Dc / BN, BNTOK / BM);
    gemm_bf16<2><<<gff2, 256, GEMM_SMEM, stream>>>(mid, w2T, b2, x2, (float*)d_out,
                                                   BNTOK, Dc, DFFc);
}
